// BaseMoeModule_19378892440175
// MI455X (gfx1250) — matrix operand order repaired and checked by reading, not compiled, not run on hardware
//
#include <hip/hip_runtime.h>
#include <hip/hip_bf16.h>

// MoE (top-2, E=8) for MI455X gfx1250.
// T=2048, H=1024, E=8, F=2048. fp32 in HBM, bf16 WMMA (f32 accumulate) via
// v_wmma_f32_16x16x32_bf16. 128x128 block tiles, BK=32, 8 wave32 waves/block,
// double-buffered LDS (global-load regs -> compute prev chunk -> LDS store).

#define TT 2048
#define HH 1024
#define FF 2048
#define EE 8

typedef __attribute__((ext_vector_type(16))) __bf16 bf16x16;
typedef __attribute__((ext_vector_type(8)))  __bf16 bf16x8;
typedef __attribute__((ext_vector_type(8)))  float  f32x8;

__device__ __forceinline__ unsigned bf16_bits(float f) {
  unsigned u = __builtin_bit_cast(unsigned, f);
  return (u + 0x7FFFu + ((u >> 16) & 1u)) >> 16;      // round-to-nearest-even
}
__device__ __forceinline__ __bf16 f32_to_bf16(float f) {
  unsigned short h = (unsigned short)bf16_bits(f);
  return __builtin_bit_cast(__bf16, h);
}
__device__ __forceinline__ unsigned pack_bf16x2(float a, float b) {
  return bf16_bits(a) | (bf16_bits(b) << 16);
}
__device__ __forceinline__ bf16x16 cat8(bf16x8 lo, bf16x8 hi) {
  return __builtin_shufflevector(lo, hi, 0,1,2,3,4,5,6,7,8,9,10,11,12,13,14,15);
}

// ---------------------------------------------------------------------------
// Init: zero loss-stat accumulators in workspace.
// ---------------------------------------------------------------------------
__global__ void moe_init(float* __restrict__ lossbuf) {
  if (threadIdx.x < 17) lossbuf[threadIdx.x] = 0.0f;
}

// ---------------------------------------------------------------------------
// Router: logits -> softmax -> top-2 (renormalized) -> dense combine [T,E],
// plus per-expert token counts, per-expert prob sums, sum(lse^2).
// ---------------------------------------------------------------------------
__global__ __launch_bounds__(256)
void moe_router(const float* __restrict__ X, const float* __restrict__ RW,
                float* __restrict__ combine, float* __restrict__ lossbuf) {
  const int tid  = threadIdx.x;
  const int lane = tid & 31;
  const int wid  = tid >> 5;
  const int t    = blockIdx.x * 8 + wid;

  float part[EE];
#pragma unroll
  for (int e = 0; e < EE; ++e) part[e] = 0.0f;

  const float* xr = X + (size_t)t * HH;
  for (int i = lane; i < HH; i += 32) {
    const float xv = xr[i];
    const float* rw = RW + (size_t)i * EE;
#pragma unroll
    for (int e = 0; e < EE; ++e) part[e] += xv * rw[e];
  }
#pragma unroll
  for (int e = 0; e < EE; ++e)
    for (int off = 16; off > 0; off >>= 1)
      part[e] += __shfl_down(part[e], off, 32);

  if (lane == 0) {
    float mx = part[0];
#pragma unroll
    for (int e = 1; e < EE; ++e) mx = fmaxf(mx, part[e]);
    float p[EE], se = 0.0f;
#pragma unroll
    for (int e = 0; e < EE; ++e) { p[e] = __expf(part[e] - mx); se += p[e]; }
    const float inv = 1.0f / se;
#pragma unroll
    for (int e = 0; e < EE; ++e) p[e] *= inv;          // softmax probs

    // top-2 (ties -> lower index, matching jax.lax.top_k)
    int i1 = 0;
#pragma unroll
    for (int e = 1; e < EE; ++e) if (p[e] > p[i1]) i1 = e;
    int i2 = (i1 == 0) ? 1 : 0;
#pragma unroll
    for (int e = 0; e < EE; ++e) if (e != i1 && p[e] > p[i2]) i2 = e;

    const float wsum = p[i1] + p[i2];
    float* cr = combine + (size_t)t * EE;
#pragma unroll
    for (int e = 0; e < EE; ++e) cr[e] = 0.0f;
    cr[i1] = p[i1] / wsum;
    cr[i2] = p[i2] / wsum;

    const float lse = mx + __logf(se);
    atomicAdd(&lossbuf[16], lse * lse);
    atomicAdd(&lossbuf[i1], 1.0f);
    atomicAdd(&lossbuf[i2], 1.0f);
#pragma unroll
    for (int e = 0; e < EE; ++e) atomicAdd(&lossbuf[8 + e], p[e]);
  }
}

// ---------------------------------------------------------------------------
// GEMM1 + silu:  mid[T,F] = bf16( silu( X[T,H] @ W1e[H,F] ) )
// Double-buffered LDS; A[m][k] and B stored transposed [n][k], bf16.
// ---------------------------------------------------------------------------
__global__ __launch_bounds__(256)
void moe_gemm1_silu(const float* __restrict__ X, const float* __restrict__ W1e,
                    __bf16* __restrict__ mid) {
  __shared__ __align__(16) __bf16 Al[2][128 * 32];
  __shared__ __align__(16) __bf16 Bl[2][128 * 32];

  const int tid  = threadIdx.x;
  const int lane = tid & 31;
  const int wid  = tid >> 5;
  const int wm   = wid >> 2;            // 0..1  (64-row slab)
  const int wn   = wid & 3;             // 0..3  (32-col slab)
  const int bm0  = blockIdx.y * 128;
  const int bn0  = blockIdx.x * 128;

  const int half = lane >> 4;
  const int l16  = lane & 15;
  const int akoff = half * 16;          // A frag: K {0..15} / {16..31}
  const int bkoff = half * 16;          // B frag: K {0..15} / {16..31}

  f32x8 acc[4][2];
#pragma unroll
  for (int mi = 0; mi < 4; ++mi)
#pragma unroll
    for (int ni = 0; ni < 2; ++ni)
#pragma unroll
      for (int r = 0; r < 8; ++r) acc[mi][ni][r] = 0.0f;

  float4 ar[4];                         // A staging regs: 4 float4 (k-quads)
  float  br[16];                        // B staging regs: 4 quads of 4 (along k)

  auto loadA = [&](int k0) {
#pragma unroll
    for (int p = 0; p < 4; ++p) {
      const int idx = tid + p * 256;            // 0..1023
      const int row = idx >> 3;                 // 0..127
      const int c4  = idx & 7;                  // k-quad 0..7
      ar[p] = *(const float4*)(X + (size_t)(bm0 + row) * HH + k0 + c4 * 4);
    }
  };
  auto loadB = [&](int k0) {
#pragma unroll
    for (int p = 0; p < 4; ++p) {
      const int idx = tid + p * 256;
      const int n   = idx & 127;                // coalesced across lanes
      const int kq  = idx >> 7;                 // 0..7
#pragma unroll
      for (int j = 0; j < 4; ++j)
        br[p * 4 + j] = W1e[(size_t)(k0 + kq * 4 + j) * FF + bn0 + n];
    }
  };
  auto storeA = [&](int buf) {
#pragma unroll
    for (int p = 0; p < 4; ++p) {
      const int idx = tid + p * 256;
      const int row = idx >> 3;
      const int c4  = idx & 7;
      uint2 v;
      v.x = pack_bf16x2(ar[p].x, ar[p].y);
      v.y = pack_bf16x2(ar[p].z, ar[p].w);
      *(uint2*)(&Al[buf][row * 32 + c4 * 4]) = v;
    }
  };
  auto storeB = [&](int buf) {
#pragma unroll
    for (int p = 0; p < 4; ++p) {
      const int idx = tid + p * 256;
      const int n   = idx & 127;
      const int kq  = idx >> 7;
      uint2 v;
      v.x = pack_bf16x2(br[p * 4 + 0], br[p * 4 + 1]);
      v.y = pack_bf16x2(br[p * 4 + 2], br[p * 4 + 3]);
      *(uint2*)(&Bl[buf][n * 32 + kq * 4]) = v;
    }
  };

  // Prologue: stage chunk 0.
  loadA(0); loadB(0);
  storeA(0); storeB(0);
  __syncthreads();

  const int NK = HH / 32;
  for (int i = 0; i < NK; ++i) {
    const int cur = i & 1;
    const bool more = (i + 1) < NK;
    if (more) { loadA((i + 1) * 32); loadB((i + 1) * 32); }  // hide under WMMA

    bf16x16 afr[4], bfr[2];
#pragma unroll
    for (int mi = 0; mi < 4; ++mi) {
      const __bf16* s = &Al[cur][(wm * 64 + mi * 16 + l16) * 32 + akoff];
      afr[mi] = cat8(*(const bf16x8*)s, *(const bf16x8*)(s + 8));
    }
#pragma unroll
    for (int ni = 0; ni < 2; ++ni) {
      const __bf16* s = &Bl[cur][(wn * 32 + ni * 16 + l16) * 32 + bkoff];
      bfr[ni] = cat8(*(const bf16x8*)s, *(const bf16x8*)(s + 8));
    }
#pragma unroll
    for (int mi = 0; mi < 4; ++mi)
#pragma unroll
      for (int ni = 0; ni < 2; ++ni)
        acc[mi][ni] = __builtin_amdgcn_wmma_f32_16x16x32_bf16(
            false, afr[mi], false, bfr[ni], (short)0, acc[mi][ni], false, false);

    if (more) { storeA(cur ^ 1); storeB(cur ^ 1); }
    __syncthreads();
  }

  // Epilogue: silu then bf16 store. C/D layout: VGPR r -> M=r (+8 for lanes>=16).
#pragma unroll
  for (int mi = 0; mi < 4; ++mi)
#pragma unroll
    for (int ni = 0; ni < 2; ++ni)
#pragma unroll
      for (int r = 0; r < 8; ++r) {
        const int grow = bm0 + wm * 64 + mi * 16 + r + half * 8;
        const int gcol = bn0 + wn * 32 + ni * 16 + l16;
        const float x = acc[mi][ni][r];
        const float s = x / (1.0f + __expf(-x));
        mid[(size_t)grow * FF + gcol] = f32_to_bf16(s);
      }
}

// ---------------------------------------------------------------------------
// GEMM2 + weighted combine:
//   out[T,H] (+)= combine[:,e] * ( mid[T,F](bf16) @ W2e[F,H] )
// ---------------------------------------------------------------------------
__global__ __launch_bounds__(256)
void moe_gemm2_combine(const __bf16* __restrict__ mid, const float* __restrict__ W2e,
                       const float* __restrict__ combine, int e, int accumulate,
                       float* __restrict__ out) {
  __shared__ __align__(16) __bf16 Al[2][128 * 32];
  __shared__ __align__(16) __bf16 Bl[2][128 * 32];
  __shared__ float cw[128];

  const int tid  = threadIdx.x;
  const int lane = tid & 31;
  const int wid  = tid >> 5;
  const int wm   = wid >> 2;
  const int wn   = wid & 3;
  const int bm0  = blockIdx.y * 128;
  const int bn0  = blockIdx.x * 128;

  const int half = lane >> 4;
  const int l16  = lane & 15;
  const int akoff = half * 16;
  const int bkoff = half * 16;

  if (tid < 128) cw[tid] = combine[(size_t)(bm0 + tid) * EE + e];

  f32x8 acc[4][2];
#pragma unroll
  for (int mi = 0; mi < 4; ++mi)
#pragma unroll
    for (int ni = 0; ni < 2; ++ni)
#pragma unroll
      for (int r = 0; r < 8; ++r) acc[mi][ni][r] = 0.0f;

  bf16x8 ar[2];                         // A staging regs (already bf16)
  float  br[16];                        // B staging regs

  auto loadA = [&](int k0) {
#pragma unroll
    for (int p = 0; p < 2; ++p) {
      const int idx = tid + p * 256;            // 0..511
      const int row = idx >> 2;                 // 0..127
      const int c8  = idx & 3;                  // 8-elem chunk 0..3
      ar[p] = *(const bf16x8*)(mid + (size_t)(bm0 + row) * FF + k0 + c8 * 8);
    }
  };
  auto loadB = [&](int k0) {
#pragma unroll
    for (int p = 0; p < 4; ++p) {
      const int idx = tid + p * 256;
      const int n   = idx & 127;
      const int kq  = idx >> 7;
#pragma unroll
      for (int j = 0; j < 4; ++j)
        br[p * 4 + j] = W2e[(size_t)(k0 + kq * 4 + j) * HH + bn0 + n];
    }
  };
  auto storeA = [&](int buf) {
#pragma unroll
    for (int p = 0; p < 2; ++p) {
      const int idx = tid + p * 256;
      const int row = idx >> 2;
      const int c8  = idx & 3;
      *(bf16x8*)(&Al[buf][row * 32 + c8 * 8]) = ar[p];
    }
  };
  auto storeB = [&](int buf) {
#pragma unroll
    for (int p = 0; p < 4; ++p) {
      const int idx = tid + p * 256;
      const int n   = idx & 127;
      const int kq  = idx >> 7;
      uint2 v;
      v.x = pack_bf16x2(br[p * 4 + 0], br[p * 4 + 1]);
      v.y = pack_bf16x2(br[p * 4 + 2], br[p * 4 + 3]);
      *(uint2*)(&Bl[buf][n * 32 + kq * 4]) = v;
    }
  };

  loadA(0); loadB(0);
  storeA(0); storeB(0);
  __syncthreads();

  const int NK = FF / 32;
  for (int i = 0; i < NK; ++i) {
    const int cur = i & 1;
    const bool more = (i + 1) < NK;
    if (more) { loadA((i + 1) * 32); loadB((i + 1) * 32); }

    bf16x16 afr[4], bfr[2];
#pragma unroll
    for (int mi = 0; mi < 4; ++mi) {
      const __bf16* s = &Al[cur][(wm * 64 + mi * 16 + l16) * 32 + akoff];
      afr[mi] = cat8(*(const bf16x8*)s, *(const bf16x8*)(s + 8));
    }
#pragma unroll
    for (int ni = 0; ni < 2; ++ni) {
      const __bf16* s = &Bl[cur][(wn * 32 + ni * 16 + l16) * 32 + bkoff];
      bfr[ni] = cat8(*(const bf16x8*)s, *(const bf16x8*)(s + 8));
    }
#pragma unroll
    for (int mi = 0; mi < 4; ++mi)
#pragma unroll
      for (int ni = 0; ni < 2; ++ni)
        acc[mi][ni] = __builtin_amdgcn_wmma_f32_16x16x32_bf16(
            false, afr[mi], false, bfr[ni], (short)0, acc[mi][ni], false, false);

    if (more) { storeA(cur ^ 1); storeB(cur ^ 1); }
    __syncthreads();
  }

#pragma unroll
  for (int mi = 0; mi < 4; ++mi)
#pragma unroll
    for (int ni = 0; ni < 2; ++ni)
#pragma unroll
      for (int r = 0; r < 8; ++r) {
        const int lrow = wm * 64 + mi * 16 + r + half * 8;
        const int grow = bm0 + lrow;
        const int gcol = bn0 + wn * 32 + ni * 16 + l16;
        const float v = cw[lrow] * acc[mi][ni][r];
        const size_t o = (size_t)grow * HH + gcol;
        if (accumulate) out[o] += v; else out[o] = v;
      }
}

// ---------------------------------------------------------------------------
// Finalize losses: LBL = 0.01*E*sum_e (count_e/T)*(probsum_e/T),
//                  RZL = 0.001*mean(lse^2)
// ---------------------------------------------------------------------------
__global__ void moe_finalize(const float* __restrict__ lossbuf,
                             float* __restrict__ out_losses) {
  if (threadIdx.x == 0) {
    float s = 0.0f;
    for (int e = 0; e < EE; ++e)
      s += (lossbuf[e] / (float)TT) * (lossbuf[8 + e] / (float)TT);
    out_losses[0] = 0.01f * (float)EE * s;
    out_losses[1] = 0.001f * (lossbuf[16] / (float)TT);
  }
}

// ---------------------------------------------------------------------------
extern "C" void kernel_launch(void* const* d_in, const int* in_sizes, int n_in,
                              void* d_out, int out_size, void* d_ws, size_t ws_size,
                              hipStream_t stream) {
  const float* X  = (const float*)d_in[0];   // [T,H]
  const float* RW = (const float*)d_in[1];   // [H,E]
  const float* W1 = (const float*)d_in[2];   // [E,H,F]
  const float* W2 = (const float*)d_in[3];   // [E,F,H]
  float* out = (float*)d_out;                // [T*H] + lbl + rzl

  float*  combine = (float*)d_ws;                          // T*E f32 = 64 KB
  float*  lossbuf = (float*)((char*)d_ws + 65536);         // 17 f32
  __bf16* mid     = (__bf16*)((char*)d_ws + 131072);       // T*F bf16 = 8 MB

  moe_init<<<1, 32, 0, stream>>>(lossbuf);
  moe_router<<<TT / 8, 256, 0, stream>>>(X, RW, combine, lossbuf);

  for (int e = 0; e < EE; ++e) {
    moe_gemm1_silu<<<dim3(FF / 128, TT / 128), 256, 0, stream>>>(
        X, W1 + (size_t)e * HH * FF, mid);
    moe_gemm2_combine<<<dim3(HH / 128, TT / 128), 256, 0, stream>>>(
        mid, W2 + (size_t)e * FF * HH, combine, e, (e > 0) ? 1 : 0, out);
  }

  moe_finalize<<<1, 1, 0, stream>>>(lossbuf, out + (size_t)TT * HH);
}
